// stsgcl_15444702396933
// MI455X (gfx1250) — matrix operand order repaired and checked by reading, not compiled, not run on hardware
//
#include <hip/hip_runtime.h>
#include <math.h>

typedef float  v8f  __attribute__((ext_vector_type(8)));
typedef __bf16 v4bf __attribute__((ext_vector_type(4)));
typedef __bf16 v8bf __attribute__((ext_vector_type(8)));
typedef __bf16 v16bf __attribute__((ext_vector_type(16)));

#define MDIM   12288   // 3N rows
#define KDIM   12288   // 3N reduction
#define NCOLS  640     // 20 (b,w) slabs * 32 channels
#define BM     128
#define BN     64
#define BK     64
#define LDAH   72      // padded bf16 LDS stride for A (conflict-free, 16B-aligned)
#define LDBH   72      // padded bf16 LDS stride for B^T
#define NT     (KDIM / BK)   // 192 k-tiles

// ---------------------------------------------------------------------------
// Kernel 1: embeddings + window gather into bf16 X[m * 640 + j]
//   m = s*4096 + n  (s in 0..2, n in 0..4095),  j = ((b*10 + w)*32 + c)
// ---------------------------------------------------------------------------
__global__ __launch_bounds__(256) void prep_kernel(
    const float* __restrict__ data, const float* __restrict__ temb,
    const float* __restrict__ semb, __bf16* __restrict__ Xh) {
  int e = blockIdx.x * 256 + threadIdx.x;           // 12288*640 = 7,864,320 total
  int j = e % NCOLS;
  int m = e / NCOLS;
  int c  = j & 31;
  int bw = j >> 5;
  int w  = bw % 10;
  int b  = bw / 10;
  int n  = m & 4095;
  int s  = m >> 12;
  int t  = w + s;
  float v = data[(((size_t)b * 12 + t) * 4096 + n) * 32 + c]
          + temb[t * 32 + c]
          + semb[n * 32 + c];
  Xh[e] = (__bf16)v;
}

// ---------------------------------------------------------------------------
// Kernel 2: Y = adj @ X  with bf16 WMMA (fp32 accumulate).
// adj is fp32 in HBM; converted to bf16 while staging into LDS.
// Block: 256 threads (8 waves), tile 128x64, BK=64, double-buffered LDS.
// Grid: (10 col-tiles, 96 row-tiles); x fastest => adj rows reused via L2.
// ---------------------------------------------------------------------------
__global__ __launch_bounds__(256) void gemm_adj(
    const float* __restrict__ adj, const __bf16* __restrict__ Xh,
    float* __restrict__ Y) {
  __shared__ __align__(16) __bf16 Ah[2][BM * LDAH];   // 2 x 18.4 KB, row-major [m][k]
  __shared__ __align__(16) __bf16 Bt[2][BN * LDBH];   // 2 x  9.2 KB, transposed [n][k]

  const int tid     = threadIdx.x;
  const int rowBase = blockIdx.y * BM;
  const int colBase = blockIdx.x * BN;

  const int lane  = tid & 31;
  const int wave  = tid >> 5;
  const int waveM = wave & 3;        // 0..3 -> 32-row strips
  const int waveN = wave >> 2;       // 0..1 -> 32-col strips
  const int half  = lane >> 4;       // lane half selects K sub-ranges (ISA layout)
  const int l15   = lane & 15;

  // staging indices
  const int aRow0 = tid >> 4;        // 0..15 (stride 16 over 128 rows -> 8 iters)
  const int aCol  = (tid & 15) * 4;  // 0..60 (fp32 cols, 4 per thread)
  const int bRow0 = tid >> 3;        // 0..31 (k, stride 32 over 64 -> 2 iters)
  const int bCol8 = (tid & 7) * 8;   // 0..56 (n, 8 bf16 per thread)

  v8f acc[2][2];
  const v8f vzero = {0.f, 0.f, 0.f, 0.f, 0.f, 0.f, 0.f, 0.f};
#pragma unroll
  for (int mi = 0; mi < 2; ++mi)
#pragma unroll
    for (int ni = 0; ni < 2; ++ni) acc[mi][ni] = vzero;

  // ---- prologue: tile kb=0 -> buffer 0 ----
#pragma unroll
  for (int r = 0; r < 8; ++r) {
    float4 v = *(const float4*)&adj[(size_t)(rowBase + aRow0 + r * 16) * KDIM + aCol];
    v4bf h;
    h[0] = (__bf16)v.x; h[1] = (__bf16)v.y; h[2] = (__bf16)v.z; h[3] = (__bf16)v.w;
    *(v4bf*)&Ah[0][(aRow0 + r * 16) * LDAH + aCol] = h;
  }
#pragma unroll
  for (int r = 0; r < 2; ++r) {
    v8bf v = *(const v8bf*)&Xh[(size_t)(bRow0 + r * 32) * NCOLS + colBase + bCol8];
#pragma unroll
    for (int i = 0; i < 8; ++i)
      Bt[0][(bCol8 + i) * LDBH + bRow0 + r * 32] = v[i];
  }
  __syncthreads();

  for (int kb = 0; kb < NT; ++kb) {
    const int cur = kb & 1;
    const bool more = (kb + 1) < NT;
    float4 ra[8];
    v8bf   rb[2];
    if (more) {
      const int k0 = (kb + 1) * BK;
#pragma unroll
      for (int r = 0; r < 8; ++r)
        ra[r] = *(const float4*)&adj[(size_t)(rowBase + aRow0 + r * 16) * KDIM + k0 + aCol];
#pragma unroll
      for (int r = 0; r < 2; ++r)
        rb[r] = *(const v8bf*)&Xh[(size_t)(k0 + bRow0 + r * 32) * NCOLS + colBase + bCol8];
    }

    // ---- compute on buffer `cur` ----
    const __bf16* __restrict__ ah = Ah[cur];
    const __bf16* __restrict__ bt = Bt[cur];
#pragma unroll
    for (int ks = 0; ks < BK; ks += 32) {       // two K=32 WMMA steps
      v16bf a[2], b[2];
#pragma unroll
      for (int mi = 0; mi < 2; ++mi) {
        const int ar = waveM * 32 + mi * 16 + l15;
        // A 16x32 in this code's contiguous k order: lane half 0 -> K 0..15; half 1 -> K 16..31 (matches B)
        v8bf lo = *(const v8bf*)&ah[ar * LDAH + ks + half * 16];
        v8bf hi = *(const v8bf*)&ah[ar * LDAH + ks + half * 16 + 8];
        a[mi] = __builtin_shufflevector(lo, hi, 0, 1, 2, 3, 4, 5, 6, 7,
                                                8, 9, 10, 11, 12, 13, 14, 15);
      }
#pragma unroll
      for (int ni = 0; ni < 2; ++ni) {
        const int bc = waveN * 32 + ni * 16 + l15;
        // B 32x16 in the same contiguous k order: lane half selects K 0..15 / 16..31, N = l15 -> contiguous in B^T
        b[ni] = *(const v16bf*)&bt[bc * LDBH + ks + half * 16];
      }
#pragma unroll
      for (int mi = 0; mi < 2; ++mi)
#pragma unroll
        for (int ni = 0; ni < 2; ++ni)
          acc[mi][ni] = __builtin_amdgcn_wmma_f32_16x16x32_bf16(
              false, a[mi], false, b[ni], (short)0, acc[mi][ni], false, false);
    }

    // ---- store next tile into the other buffer ----
    if (more) {
      const int nxt = cur ^ 1;
#pragma unroll
      for (int r = 0; r < 8; ++r) {
        v4bf h;
        h[0] = (__bf16)ra[r].x; h[1] = (__bf16)ra[r].y;
        h[2] = (__bf16)ra[r].z; h[3] = (__bf16)ra[r].w;
        *(v4bf*)&Ah[nxt][(aRow0 + r * 16) * LDAH + aCol] = h;
      }
#pragma unroll
      for (int r = 0; r < 2; ++r)
#pragma unroll
        for (int i = 0; i < 8; ++i)
          Bt[nxt][(bCol8 + i) * LDBH + bRow0 + r * 32] = rb[r][i];
    }
    __syncthreads();
  }

  // ---- epilogue: C/D layout => VGPR r holds row (r + 8*half), col l15 ----
#pragma unroll
  for (int mi = 0; mi < 2; ++mi) {
#pragma unroll
    for (int ni = 0; ni < 2; ++ni) {
      const int row = rowBase + waveM * 32 + mi * 16 + half * 8;
      const int col = colBase + waveN * 32 + ni * 16 + l15;
      float* yp = Y + (size_t)row * NCOLS + col;
#pragma unroll
      for (int r = 0; r < 8; ++r) yp[(size_t)r * NCOLS] = acc[mi][ni][r];
    }
  }
}

// ---------------------------------------------------------------------------
// Kernel 3: per (m, bw): z = y(1x32) @ W(32x64) + b ; GLU ; write next X (bf16);
// for middle rows also store/max into out (fp32).
// ---------------------------------------------------------------------------
__global__ __launch_bounds__(256) void glu_kernel(
    const float* __restrict__ Y, const float* __restrict__ W,
    const float* __restrict__ bias, __bf16* __restrict__ Xn,
    float* __restrict__ out, int layer) {
  __shared__ float Ws[32 * 64];
  __shared__ float bs[64];
  for (int i = threadIdx.x; i < 2048; i += 256) Ws[i] = W[i];
  if (threadIdx.x < 64) bs[threadIdx.x] = bias[threadIdx.x];
  __syncthreads();

  const int idx = blockIdx.x * 256 + threadIdx.x;   // 12288*20 = 245,760
  const int bw = idx % 20;
  const int m  = idx / 20;

  float y[32];
  const float* yp = Y + (size_t)m * NCOLS + bw * 32;
#pragma unroll
  for (int c = 0; c < 32; ++c) y[c] = yp[c];

  float xo[32];
#pragma unroll 4
  for (int co = 0; co < 32; ++co) {
    float lhs = bs[co];
    float rhs = bs[co + 32];
#pragma unroll
    for (int c = 0; c < 32; ++c) {
      lhs = fmaf(y[c], Ws[c * 64 + co], lhs);
      rhs = fmaf(y[c], Ws[c * 64 + co + 32], rhs);
    }
    xo[co] = lhs * (1.0f / (1.0f + __expf(-rhs)));
  }

  __bf16* xp = Xn + (size_t)m * NCOLS + bw * 32;
#pragma unroll
  for (int co = 0; co < 32; ++co) xp[co] = (__bf16)xo[co];

  if (m >= 4096 && m < 8192) {
    float* op = out + ((size_t)bw * 4096 + (m - 4096)) * 32;
    if (layer == 0) {
#pragma unroll
      for (int co = 0; co < 32; ++co) op[co] = xo[co];
    } else {
#pragma unroll
      for (int co = 0; co < 32; ++co) op[co] = fmaxf(op[co], xo[co]);
    }
  }
}

// ---------------------------------------------------------------------------
extern "C" void kernel_launch(void* const* d_in, const int* in_sizes, int n_in,
                              void* d_out, int out_size, void* d_ws, size_t ws_size,
                              hipStream_t stream) {
  const float* data = (const float*)d_in[0];
  const float* adj  = (const float*)d_in[1];
  const float* temb = (const float*)d_in[2];
  const float* semb = (const float*)d_in[3];
  const float* Wl[3] = {(const float*)d_in[4], (const float*)d_in[6], (const float*)d_in[8]};
  const float* bl[3] = {(const float*)d_in[5], (const float*)d_in[7], (const float*)d_in[9]};
  float* out = (float*)d_out;

  // workspace: Xh (bf16, 15.7 MB) at 0, Y (fp32, 31.5 MB) at +16 MB
  __bf16* Xh = (__bf16*)d_ws;
  float*  Y  = (float*)((char*)d_ws + (size_t)(16u << 20));

  prep_kernel<<<(MDIM * NCOLS) / 256, 256, 0, stream>>>(data, temb, semb, Xh);

  for (int l = 0; l < 3; ++l) {
    gemm_adj<<<dim3(NCOLS / BN, MDIM / BM), 256, 0, stream>>>(adj, Xh, Y);
    glu_kernel<<<(MDIM * 20) / 256, 256, 0, stream>>>(Y, Wl[l], bl[l], Xh, out, l);
  }
}
